// MultiHeadAttentionReciprocal_17179869715
// MI455X (gfx1250) — matrix operand order repaired and checked by reading, not compiled, not run on hardware
//
#include <hip/hip_runtime.h>
#include <hip/hip_bf16.h>

typedef __attribute__((ext_vector_type(16))) _Float16 v16h;
typedef __attribute__((ext_vector_type(8)))  _Float16 v8h;
typedef __attribute__((ext_vector_type(8)))  float    v8f;

union V16U { v16h v; v8h h[2]; };

#define HEADS 8
#define DMODEL 128
#define BSZ 4
#define LQ 1024
#define LK 2048

// ---------------------------------------------------------------------------
// WMMA fragment helpers (CDNA5 16x16x32 f16, wave32 layouts per ISA 7.12.2).
// A (16x32, MxK): lane m=lane&15, hi=lane>>4:
//   a[0..7] = row m, K = k0+8*hi+0..7 ; a[8..15] = row m, K = k0+16+8*hi+0..7
//   -> two contiguous 16B runs -> two b128 loads.
// B (32x16, KxN) from a TRANSPOSED [N][K] buffer: lane n=lane&15:
//   b[0..7] = Bt[n][k0+8*hi+0..7] ; b[8..15] = Bt[n][k0+16+8*hi+0..7] -> two b128 loads.
// C/D f32 16x16: lane n=lane&15; VGPR r holds M = r + 8*hi.
// ---------------------------------------------------------------------------
static __device__ __forceinline__ v16h load_a_frag(const _Float16* __restrict__ A,
                                                   int lda, int k0, int lane) {
  int m = lane & 15, hi = lane >> 4;
  const _Float16* row = A + (size_t)m * lda + k0 + 8 * hi;
  V16U u;
  u.h[0] = *(const v8h*)(row);
  u.h[1] = *(const v8h*)(row + 16);
  return u.v;
}

static __device__ __forceinline__ v16h load_bT_frag(const _Float16* __restrict__ Bt,
                                                    int ldt, int k0, int lane) {
  int n = lane & 15, hi = lane >> 4;
  const _Float16* col = Bt + (size_t)n * ldt + k0 + 8 * hi;
  V16U u;
  u.h[0] = *(const v8h*)(col);
  u.h[1] = *(const v8h*)(col + 16);
  return u.v;
}

// ---------------------------------------------------------------------------
// 0) convert the 8 weight matrices (128x128 f32) to f16 TRANSPOSED: Wt[n][k]
// ---------------------------------------------------------------------------
struct WPtrs { const float* p[8]; };

__global__ __launch_bounds__(256) void cvt_weights_kernel(WPtrs w, _Float16* __restrict__ dst) {
  int i = blockIdx.x * 256 + threadIdx.x;          // 0 .. 8*16384-1
  int m = i >> 14, o = i & 16383;
  int k = o >> 7, n = o & 127;
  dst[(m << 14) + n * 128 + k] = (_Float16)w.p[m][o];
}

// ---------------------------------------------------------------------------
// 1a) projection (row-major out): Out[Mx128 f16] = X[Mx128 f32] @ W + b
//     block = 256 (8 waves) covers 16 rows x 128 cols; Wt staged in LDS.
// ---------------------------------------------------------------------------
__global__ __launch_bounds__(256)
void proj_kernel(const float* __restrict__ X, const _Float16* __restrict__ Wt,
                 const float* __restrict__ bias, _Float16* __restrict__ Out) {
  __shared__ _Float16 ldsX[16 * 128];
  __shared__ _Float16 ldsWt[128 * 128];
  int tid = threadIdx.x, lane = tid & 31, wave = tid >> 5;
  int row0 = blockIdx.x * 16;

  for (int i = tid; i < 16 * 128; i += 256)
    ldsX[i] = (_Float16)X[(size_t)(row0 + (i >> 7)) * 128 + (i & 127)];
  for (int i = tid; i < 2048; i += 256)            // 128x128 halves as int4
    ((int4*)ldsWt)[i] = ((const int4*)Wt)[i];
  __syncthreads();

  int n = lane & 15, hi = lane >> 4;
  v8f acc = {};
#pragma unroll
  for (int kk = 0; kk < 4; ++kk) {
    v16h a = load_a_frag(ldsX, 128, kk * 32, lane);
    v16h b = load_bT_frag(ldsWt + (size_t)wave * 16 * 128, 128, kk * 32, lane);
    acc = __builtin_amdgcn_wmma_f32_16x16x32_f16(false, a, false, b, (short)0, acc, false, false);
  }
  float bv = bias[wave * 16 + n];
#pragma unroll
  for (int r = 0; r < 8; ++r)
    Out[(size_t)(row0 + r + 8 * hi) * 128 + wave * 16 + n] = (_Float16)(acc[r] + bv);
}

// ---------------------------------------------------------------------------
// 1b) projection with TRANSPOSED output: OutT[(b*128+col)][k] (col = h*16+d).
//     Per-lane result is 8 consecutive k -> a single b128 store.
// ---------------------------------------------------------------------------
__global__ __launch_bounds__(256)
void proj_kernel_t(const float* __restrict__ X, const _Float16* __restrict__ Wt,
                   const float* __restrict__ bias, _Float16* __restrict__ OutT, int Krows) {
  __shared__ _Float16 ldsX[16 * 128];
  __shared__ _Float16 ldsWt[128 * 128];
  int tid = threadIdx.x, lane = tid & 31, wave = tid >> 5;
  int row0 = blockIdx.x * 16;

  for (int i = tid; i < 16 * 128; i += 256)
    ldsX[i] = (_Float16)X[(size_t)(row0 + (i >> 7)) * 128 + (i & 127)];
  for (int i = tid; i < 2048; i += 256)
    ((int4*)ldsWt)[i] = ((const int4*)Wt)[i];
  __syncthreads();

  int n = lane & 15, hi = lane >> 4;
  v8f acc = {};
#pragma unroll
  for (int kk = 0; kk < 4; ++kk) {
    v16h a = load_a_frag(ldsX, 128, kk * 32, lane);
    v16h b = load_bT_frag(ldsWt + (size_t)wave * 16 * 128, 128, kk * 32, lane);
    acc = __builtin_amdgcn_wmma_f32_16x16x32_f16(false, a, false, b, (short)0, acc, false, false);
  }
  float bv = bias[wave * 16 + n];
  int col = wave * 16 + n;
  int b = row0 / Krows;
  int kloc = row0 - b * Krows + 8 * hi;
  union { v8h v; _Float16 e[8]; } o;
#pragma unroll
  for (int r = 0; r < 8; ++r) o.e[r] = (_Float16)(acc[r] + bv);
  *(v8h*)(OutT + ((size_t)b * 128 + col) * Krows + kloc) = o.v;
}

// ---------------------------------------------------------------------------
// 2) scores: S = (Q_h @ K_h^T)/4 into A [b,h,q,k] and transposed A2 [b,h,k,q].
//    K-dim = 16: hi=1 halves of both fragments are structurally zero.
//    grid = (LQ/32, LK/64, B*H); 8 waves -> 2x4 grid of 16x16 WMMA tiles.
// ---------------------------------------------------------------------------
__global__ __launch_bounds__(256)
void scores_kernel(const _Float16* __restrict__ Qh, const _Float16* __restrict__ Kh,
                   float* __restrict__ A, float* __restrict__ A2) {
  __shared__ _Float16 ldsQ[32 * 16];
  __shared__ _Float16 ldsK[64 * 16];
  int tid = threadIdx.x, lane = tid & 31, wave = tid >> 5;
  int bh = blockIdx.z, b = bh >> 3, h = bh & 7;
  int q0 = blockIdx.x * 32, k0 = blockIdx.y * 64;
  const _Float16* Qbase = Qh + (size_t)b * LQ * 128 + h * 16;
  const _Float16* Kbase = Kh + (size_t)b * LK * 128 + h * 16;

  for (int i = tid; i < 64; i += 256)
    ((int4*)ldsQ)[i] = *(const int4*)(Qbase + (size_t)(q0 + (i >> 1)) * 128 + (i & 1) * 8);
  for (int i = tid; i < 128; i += 256)
    ((int4*)ldsK)[i] = *(const int4*)(Kbase + (size_t)(k0 + (i >> 1)) * 128 + (i & 1) * 8);
  __syncthreads();

  int qi = wave & 1, ki = wave >> 1;
  int n = lane & 15, hi = lane >> 4, m = lane & 15;
  const v8h z8 = {0, 0, 0, 0, 0, 0, 0, 0};

  V16U ua;                                          // a[j]=K j (j<16) for hi=0, hi=1 -> 0
  ua.h[0] = hi == 0 ? *(const v8h*)(ldsQ + (qi * 16 + m) * 16) : z8;
  ua.h[1] = hi == 0 ? *(const v8h*)(ldsQ + (qi * 16 + m) * 16 + 8) : z8;
  V16U ub;                                          // b[j]=Kt row n, j<16, hi=1 -> 0
  if (hi == 0) {
    const _Float16* kr = ldsK + (ki * 16 + n) * 16;
    ub.h[0] = *(const v8h*)(kr);
    ub.h[1] = *(const v8h*)(kr + 8);
  } else {
    ub.h[0] = z8; ub.h[1] = z8;
  }
  v8f acc = {};
  acc = __builtin_amdgcn_wmma_f32_16x16x32_f16(false, ua.v, false, ub.v, (short)0, acc, false, false);

  const float scale = 0.25f;                        // 1/sqrt(d_k)
  size_t qg = (size_t)q0 + qi * 16;
  size_t kg = (size_t)k0 + ki * 16 + n;
#pragma unroll
  for (int r = 0; r < 8; ++r)                       // streaming NT stores
    __builtin_nontemporal_store(acc[r] * scale,
        A + ((size_t)bh * LQ + qg + r + 8 * hi) * LK + kg);
  float* a2p = A2 + ((size_t)bh * LK + kg) * LQ + qg + 8 * hi;
  *(float4*)(a2p)     = make_float4(acc[0] * scale, acc[1] * scale, acc[2] * scale, acc[3] * scale);
  *(float4*)(a2p + 4) = make_float4(acc[4] * scale, acc[5] * scale, acc[6] * scale, acc[7] * scale);
}

// ---------------------------------------------------------------------------
// 3) exact 1.5-entmax over rows, in place. One 256-thread block per row.
//    LDS bitonic sort (descending) + chunked cumsums + tau array + support.
// ---------------------------------------------------------------------------
__global__ __launch_bounds__(256)
void entmax_kernel(float* __restrict__ P, int K) {
  __shared__ float sX[2048];
  __shared__ float sS[2048];
  __shared__ float sT[2048];
  __shared__ float rf[256], rg[256];
  __shared__ int   ri[256];
  int tid = threadIdx.x;
  float* p = P + (size_t)blockIdx.x * K;

  float mx = -3.4e38f;
  for (int i = tid; i < K; i += 256) { float v = 0.5f * p[i]; sX[i] = v; mx = fmaxf(mx, v); }
  rf[tid] = mx; __syncthreads();
  for (int s = 128; s > 0; s >>= 1) { if (tid < s) rf[tid] = fmaxf(rf[tid], rf[tid + s]); __syncthreads(); }
  mx = rf[0]; __syncthreads();
  for (int i = tid; i < K; i += 256) { float v = sX[i] - mx; sX[i] = v; sS[i] = v; }
  __syncthreads();

  for (int k2 = 2; k2 <= K; k2 <<= 1) {
    for (int j2 = k2 >> 1; j2 > 0; j2 >>= 1) {
      for (int i = tid; i < K; i += 256) {
        int ixj = i ^ j2;
        if (ixj > i) {
          float va = sS[i], vb = sS[ixj];
          bool sw = ((i & k2) == 0) ? (va < vb) : (va > vb);
          if (sw) { sS[i] = vb; sS[ixj] = va; }
        }
      }
      __syncthreads();
    }
  }

  int C = K >> 8, base = tid * C;
  float s1 = 0.f, s2 = 0.f;
  for (int j = 0; j < C; ++j) { float v = sS[base + j]; s1 += v; s2 += v * v; }
  rf[tid] = s1; rg[tid] = s2; __syncthreads();
  if (tid == 0) {
    float a0 = 0.f, b0 = 0.f;
    for (int t = 0; t < 256; ++t) { float x = rf[t], y = rg[t]; rf[t] = a0; rg[t] = b0; a0 += x; b0 += y; }
  }
  __syncthreads();

  float c1 = rf[tid], c2 = rg[tid];
  int cnt = 0;
  for (int j = 0; j < C; ++j) {
    int i = base + j;
    float v = sS[i];
    c1 += v; c2 += v * v;
    float rho = (float)(i + 1);
    float mean = c1 / rho, meansq = c2 / rho;
    float delta = fmaxf((1.0f - rho * (meansq - mean * mean)) / rho, 0.0f);
    float tau = mean - sqrtf(delta);
    sT[i] = tau;
    cnt += (tau <= v) ? 1 : 0;
  }
  ri[tid] = cnt; __syncthreads();
  for (int s = 128; s > 0; s >>= 1) { if (tid < s) ri[tid] += ri[tid + s]; __syncthreads(); }
  int support = ri[0] > 0 ? ri[0] : 1;
  float tau_star = sT[support - 1];
  for (int i = tid; i < K; i += 256) {
    float v = fmaxf(sX[i] - tau_star, 0.0f);
    p[i] = v * v;
  }
}

// ---------------------------------------------------------------------------
// 4) AV: Out = A[f32, MxK] @ V[f16, Kx16] per (b,h). V is pre-transposed
//    [(b*128+col)][k] so LDS staging is a straight int4 copy and B-fragments
//    are two b128 DS loads. A converted f32->f16 via aligned float4 loads.
// ---------------------------------------------------------------------------
__global__ __launch_bounds__(256)
void av_kernel(const float* __restrict__ A_all, const _Float16* __restrict__ Vt_all,
               _Float16* __restrict__ O_all, int M, int K) {
  __shared__ _Float16 ldsVt[16 * 256];              // [n][256] chunk, 8KB
  int tid = threadIdx.x, lane = tid & 31, wave = tid >> 5;
  int bh = blockIdx.z, b = bh >> 3, h = bh & 7;
  const float*    Am = A_all  + (size_t)bh * M * K;
  const _Float16* Vt = Vt_all + ((size_t)b * 128 + h * 16) * K;
  _Float16*       Om = O_all  + (size_t)b * M * 128 + h * 16;
  int q0 = blockIdx.x * 128 + wave * 16;
  int n = lane & 15, hi = lane >> 4, m = lane & 15;
  v8f acc = {};

  for (int kc = 0; kc < K; kc += 256) {
    __syncthreads();
    for (int i = tid; i < 512; i += 256)            // 16 cols x 32 int4
      ((int4*)ldsVt)[i] = ((const int4*)(Vt + (size_t)(i >> 5) * K + kc))[i & 31];
    __syncthreads();
    if (kc + 256 < K)                               // global_prefetch_b8 next chunk
      __builtin_prefetch(Am + (size_t)(q0 + m) * K + kc + 256, 0, 0);
#pragma unroll
    for (int kk = 0; kk < 8; ++kk) {
      const float* arow = Am + (size_t)(q0 + m) * K + kc + kk * 32;
      float4 x0 = *(const float4*)(arow + 8 * hi);
      float4 x1 = *(const float4*)(arow + 8 * hi + 4);
      float4 x2 = *(const float4*)(arow + 16 + 8 * hi);
      float4 x3 = *(const float4*)(arow + 16 + 8 * hi + 4);
      v16h a;
      a[0]  = (_Float16)x0.x; a[1]  = (_Float16)x0.y; a[2]  = (_Float16)x0.z; a[3]  = (_Float16)x0.w;
      a[4]  = (_Float16)x1.x; a[5]  = (_Float16)x1.y; a[6]  = (_Float16)x1.z; a[7]  = (_Float16)x1.w;
      a[8]  = (_Float16)x2.x; a[9]  = (_Float16)x2.y; a[10] = (_Float16)x2.z; a[11] = (_Float16)x2.w;
      a[12] = (_Float16)x3.x; a[13] = (_Float16)x3.y; a[14] = (_Float16)x3.z; a[15] = (_Float16)x3.w;
      v16h bf = load_bT_frag(ldsVt, 256, kk * 32, lane);
      acc = __builtin_amdgcn_wmma_f32_16x16x32_f16(false, a, false, bf, (short)0, acc, false, false);
    }
  }
#pragma unroll
  for (int r = 0; r < 8; ++r)
    Om[(size_t)(q0 + r + 8 * hi) * 128 + n] = (_Float16)acc[r];
}

// ---------------------------------------------------------------------------
// 5) epilogue: relu(layernorm(Xpre@W1 + b1 + Resid@W2)). W transposed in ws.
// ---------------------------------------------------------------------------
__global__ __launch_bounds__(256)
void epilogue_kernel(const _Float16* __restrict__ Xpre, const float* __restrict__ Resid,
                     const _Float16* __restrict__ W1t, const float* __restrict__ b1,
                     const _Float16* __restrict__ W2t,
                     const float* __restrict__ gamma, const float* __restrict__ beta,
                     float* __restrict__ Out) {
  __shared__ _Float16 ldsA[16 * 128];
  __shared__ _Float16 ldsR[16 * 128];
  __shared__ float sT[16 * 128];
  __shared__ float r1[256], r2[256];
  int tid = threadIdx.x, lane = tid & 31, wave = tid >> 5;
  int row0 = blockIdx.x * 16;

  for (int i = tid; i < 256; i += 256)
    ((int4*)ldsA)[i] = ((const int4*)(Xpre + (size_t)row0 * 128))[i];
  for (int i = tid; i < 2048; i += 256)
    ldsR[i] = (_Float16)Resid[(size_t)row0 * 128 + i];
  __syncthreads();

  int n = lane & 15, hi = lane >> 4;
  v8f acc = {};
#pragma unroll
  for (int kk = 0; kk < 4; ++kk) {
    v16h a  = load_a_frag(ldsA, 128, kk * 32, lane);
    v16h bf = load_bT_frag(W1t + (size_t)wave * 16 * 128, 128, kk * 32, lane);
    acc = __builtin_amdgcn_wmma_f32_16x16x32_f16(false, a, false, bf, (short)0, acc, false, false);
  }
#pragma unroll
  for (int kk = 0; kk < 4; ++kk) {
    v16h a  = load_a_frag(ldsR, 128, kk * 32, lane);
    v16h bf = load_bT_frag(W2t + (size_t)wave * 16 * 128, 128, kk * 32, lane);
    acc = __builtin_amdgcn_wmma_f32_16x16x32_f16(false, a, false, bf, (short)0, acc, false, false);
  }
  float bv = b1[wave * 16 + n];
#pragma unroll
  for (int r = 0; r < 8; ++r)
    sT[(r + 8 * hi) * 128 + wave * 16 + n] = acc[r] + bv;
  __syncthreads();

  int rr = tid >> 4, c0 = tid & 15;
  float s = 0.f, q2 = 0.f;
#pragma unroll
  for (int j = 0; j < 8; ++j) { float v = sT[rr * 128 + c0 + 16 * j]; s += v; q2 += v * v; }
  r1[tid] = s; r2[tid] = q2; __syncthreads();
  if (c0 == 0) {
    float ss = 0.f, qq = 0.f;
    for (int t = 0; t < 16; ++t) { ss += r1[tid + t]; qq += r2[tid + t]; }
    float mu = ss * (1.0f / 128.0f);
    float var = qq * (1.0f / 128.0f) - mu * mu;
    r1[tid] = mu;
    r2[tid] = rsqrtf(fmaxf(var, 0.0f) + 1e-5f);
  }
  __syncthreads();
  float mu = r1[rr * 16], rstd = r2[rr * 16];
#pragma unroll
  for (int j = 0; j < 8; ++j) {
    int c = c0 + 16 * j;
    float v = (sT[rr * 128 + c] - mu) * rstd * gamma[c] + beta[c];
    Out[(size_t)(row0 + rr) * 128 + c] = fmaxf(v, 0.0f);
  }
}

// ---------------------------------------------------------------------------
extern "C" void kernel_launch(void* const* d_in, const int* in_sizes, int n_in,
                              void* d_out, int out_size, void* d_ws, size_t ws_size,
                              hipStream_t stream) {
  const float* q    = (const float*)d_in[0];
  const float* k    = (const float*)d_in[1];
  const float* v    = (const float*)d_in[2];
  const float* v_2  = (const float*)d_in[3];
  const float* W_Q  = (const float*)d_in[4];
  const float* b_Q  = (const float*)d_in[5];
  const float* W_K  = (const float*)d_in[6];
  const float* b_K  = (const float*)d_in[7];
  const float* W_V  = (const float*)d_in[8];
  const float* b_V  = (const float*)d_in[9];
  const float* W_V2 = (const float*)d_in[10];
  const float* b_V2 = (const float*)d_in[11];
  const float* W_O  = (const float*)d_in[12];
  const float* b_O  = (const float*)d_in[13];
  const float* W_O2 = (const float*)d_in[14];
  const float* b_O2 = (const float*)d_in[15];
  const float* W_R  = (const float*)d_in[16];
  const float* W_R2 = (const float*)d_in[17];
  const float* gamma = (const float*)d_in[18];
  const float* beta  = (const float*)d_in[19];

  float* out  = (float*)d_out;                         // [4,1024,128]
  float* out2 = out + (size_t)BSZ * LQ * DMODEL;       // [4,2048,128]
  float* A    = out2 + (size_t)BSZ * LK * DMODEL;      // [4,8,1024,2048]
  float* A2   = A + (size_t)BSZ * HEADS * LQ * LK;     // [4,8,2048,1024]

  char* ws = (char*)d_ws;
  _Float16* wgtT = (_Float16*)ws;                      // 8 transposed f16 128x128
  _Float16* hQ   = (_Float16*)(ws + 8 * 16384 * sizeof(_Float16));
  _Float16* hK   = hQ   + (size_t)BSZ * LQ * DMODEL;
  _Float16* hVt  = hK   + (size_t)BSZ * LK * DMODEL;   // V transposed  [(b*128+c)][LK]
  _Float16* hV2t = hVt  + (size_t)BSZ * LK * DMODEL;   // V2 transposed [(b*128+c)][LQ]
  _Float16* hO1  = hV2t + (size_t)BSZ * LQ * DMODEL;   // out  pre-epilogue
  _Float16* hO2  = hO1  + (size_t)BSZ * LQ * DMODEL;   // out2 pre-epilogue

  WPtrs wp{{W_Q, W_K, W_V, W_V2, W_O, W_O2, W_R, W_R2}};
  cvt_weights_kernel<<<512, 256, 0, stream>>>(wp, wgtT);

  proj_kernel<<<BSZ * LQ / 16, 256, 0, stream>>>(q, wgtT + 0 * 16384, b_Q, hQ);
  proj_kernel<<<BSZ * LK / 16, 256, 0, stream>>>(k, wgtT + 1 * 16384, b_K, hK);
  proj_kernel_t<<<BSZ * LK / 16, 256, 0, stream>>>(v,   wgtT + 2 * 16384, b_V,  hVt,  LK);
  proj_kernel_t<<<BSZ * LQ / 16, 256, 0, stream>>>(v_2, wgtT + 3 * 16384, b_V2, hV2t, LQ);

  scores_kernel<<<dim3(LQ / 32, LK / 64, BSZ * HEADS), 256, 0, stream>>>(hQ, hK, A, A2);

  entmax_kernel<<<BSZ * HEADS * LQ, 256, 0, stream>>>(A,  LK);
  entmax_kernel<<<BSZ * HEADS * LK, 256, 0, stream>>>(A2, LQ);

  av_kernel<<<dim3(LQ / 128, 1, BSZ * HEADS), 256, 0, stream>>>(A,  hVt,  hO1, LQ, LK);
  av_kernel<<<dim3(LK / 128, 1, BSZ * HEADS), 256, 0, stream>>>(A2, hV2t, hO2, LK, LQ);

  epilogue_kernel<<<BSZ * LQ / 16, 256, 0, stream>>>(hO1, q, wgtT + 4 * 16384, b_O,
                                                     wgtT + 6 * 16384, gamma, beta, out);
  epilogue_kernel<<<BSZ * LK / 16, 256, 0, stream>>>(hO2, k, wgtT + 5 * 16384, b_O2,
                                                     wgtT + 7 * 16384, gamma, beta, out2);
}
